// Seq2seq_84499186581953
// MI455X (gfx1250) — matrix operand order repaired and checked by reading, not compiled, not run on hardware
//
#include <hip/hip_runtime.h>
#include <stdint.h>

#define B_    128
#define T_    256
#define V_    512
#define H_    1024
#define FOURH 4096
#define EOS_  1

// dense kernel A-tile LDS stride (elements): 2064B == 16 (mod 256) -> conflict-free b128
#define LDS_STRIDE 1032

// lstm kernel K-blocked B-panel layout (all sizes in elements):
//   per K-chunk: 64 rows x 32 K-values, row padded 32 -> 40 elems (80B == 16 mod 64)
#define ROW_ELEMS   40                    // padded row within a K-chunk
#define KB_ELEMS    (64 * ROW_ELEMS)      // 2560 elems = 5120B per K-chunk
#define GATE_ELEMS  (16 * ROW_ELEMS)      // 640 elems between gate groups

typedef __attribute__((ext_vector_type(16))) __bf16 v16bf;
typedef __attribute__((ext_vector_type(8)))  float  v8f;
typedef __attribute__((ext_vector_type(4)))  int    v4i_;

union V16U { v16bf v; uint4 q[2]; };
union V8U  { v8f   v; float f[8]; };

#if __has_builtin(__builtin_amdgcn_global_load_async_to_lds_b128) && \
    __has_builtin(__builtin_amdgcn_s_wait_asynccnt)
#define HAVE_ASYNC_LDS 1
#else
#define HAVE_ASYNC_LDS 0
#endif

__device__ __forceinline__ void lds_cp16(const unsigned short* gsrc, unsigned short* ldst) {
#if HAVE_ASYNC_LDS
    __builtin_amdgcn_global_load_async_to_lds_b128(
        (__attribute__((address_space(1))) v4i_*)gsrc,
        (__attribute__((address_space(3))) v4i_*)ldst, 0, 0);
#else
    *(uint4*)ldst = *(const uint4*)gsrc;   // sync fallback: global->VGPR->ds_store
#endif
}
__device__ __forceinline__ void lds_cp_wait() {
#if HAVE_ASYNC_LDS
    __builtin_amdgcn_s_wait_asynccnt(0);
#endif
}

__device__ __forceinline__ unsigned short f2bf(float x) {
    union { float f; unsigned u; } v; v.f = x;
    unsigned r = v.u + 0x7FFFu + ((v.u >> 16) & 1u);   // round-to-nearest-even
    return (unsigned short)(r >> 16);
}
__device__ __forceinline__ float sigmoidf_(float x) { return 1.0f / (1.0f + __expf(-x)); }

// ---------------------------------------------------------------------------
// Token extraction: one-hot [rows, V] -> int token ids
// ---------------------------------------------------------------------------
__global__ void tokenize_kernel(const float* __restrict__ x, int* __restrict__ tok, int nRows) {
    int i = blockIdx.x * blockDim.x + threadIdx.x;
    if (i >= nRows) return;
    const float* row = x + (size_t)i * V_;
    int tv = 0;
    for (int v = 0; v < V_; v += 4) {
        float4 f = *(const float4*)(row + v);
        if (f.x > 0.5f) tv = v;
        if (f.y > 0.5f) tv = v + 1;
        if (f.z > 0.5f) tv = v + 2;
        if (f.w > 0.5f) tv = v + 3;
    }
    tok[i] = tv;
}

// idx[b] = first t with token==EOS, else T-1   (== lengths-1 in the reference)
__global__ void seqidx_kernel(const int* __restrict__ enc_tok, int* __restrict__ idx) {
    int b = threadIdx.x;
    if (b >= B_) return;
    int e = T_ - 1;
    for (int t = 0; t < T_; ++t) {
        if (enc_tok[b * T_ + t] == EOS_) { e = t; break; }
    }
    idx[b] = e;
}

// ---------------------------------------------------------------------------
// f32 [R,C] row-major  ->  bf16 [C,R] row-major (transposed), tiled via LDS.
// ---------------------------------------------------------------------------
__global__ void transpose_f32_to_bf16(const float* __restrict__ src,
                                      unsigned short* __restrict__ dst, int R, int C) {
    __shared__ unsigned short tile[32][33];
    int c0 = blockIdx.x * 32, r0 = blockIdx.y * 32;
#pragma unroll
    for (int j = 0; j < 4; ++j) {
        int r  = r0 + threadIdx.y + j * 8;
        int cc = c0 + threadIdx.x;
        tile[threadIdx.y + j * 8][threadIdx.x] = f2bf(src[(size_t)r * C + cc]);
    }
    __syncthreads();
#pragma unroll
    for (int j = 0; j < 4; ++j) {
        int cc = c0 + threadIdx.y + j * 8;
        int r  = r0 + threadIdx.x;
        dst[(size_t)cc * R + r] = tile[threadIdx.x][threadIdx.y + j * 8];
    }
}

__global__ void init_state_kernel(float* __restrict__ c, unsigned short* __restrict__ h, int n) {
    int i = blockIdx.x * blockDim.x + threadIdx.x;
    if (i < n) { c[i] = 0.0f; h[i] = (unsigned short)0; }
}

__global__ void copy_state_kernel(const float* __restrict__ csel,
                                  const unsigned short* __restrict__ hsel,
                                  float* __restrict__ c, unsigned short* __restrict__ h, int n) {
    int i = blockIdx.x * blockDim.x + threadIdx.x;
    if (i < n) { c[i] = csel[i]; h[i] = hsel[i]; }
}

// ---------------------------------------------------------------------------
// Fused LSTM step:  z = Wi[tok] + h_prev @ Wh + b ; gates ; c,h update.
// WhT: bf16 [4H, H] (pre-transposed).  Grid: H/16 blocks x 256 threads.
// Block stages its full B panel (4 gates x 16 cols x K=1024 bf16 = 128KB) in
// LDS via async-to-LDS using a K-blocked layout so every ds_load in the K-loop
// is a small immediate offset off one per-lane base pointer.  Loads are split
// from WMMAs so all 4 B fragments live in distinct registers (deep DS pipe).
// ---------------------------------------------------------------------------
__global__ void __launch_bounds__(256) lstm_step_kernel(
    const unsigned short* __restrict__ WhT,   // [4H, H] bf16
    const float*          __restrict__ Wi,    // [V, 4H] f32
    const float*          __restrict__ bias,  // [4H]
    const int*            __restrict__ tok,   // [B, tokStride]
    int tokStride, int t,
    const unsigned short* __restrict__ hprev, // [B, H] bf16
    float*                __restrict__ cbuf,  // [B, H] f32 (in-place per element)
    unsigned short*       __restrict__ hnext, // [B, H] bf16
    const int*            __restrict__ idx,   // encoder: [B] else null
    float*                __restrict__ csel,  // encoder selected c
    unsigned short*       __restrict__ hsel,  // encoder selected h
    unsigned short*       __restrict__ hd,    // decoder: [B*Tdec, H] else null
    int Tdec)
{
    extern __shared__ unsigned short smemB[];   // 32 K-chunks x 64 rows x ROW_ELEMS

    const int lane = threadIdx.x & 31;
    const int wave = threadIdx.x >> 5;   // 0..7  -> M tile
    const int half = lane >> 4;          // 0/1   -> lane group
    const int l16  = lane & 15;
    const int m0   = wave * 16;
    const int n0   = blockIdx.x * 16;

    // ---- stage B panel into LDS, K-blocked ----
    // 8192 b128 chunks: c -> kb = c>>8, rr = (c>>2)&63, cc = c&3
    // (groups of 4 threads read 64B contiguous from one row of WhT)
#pragma unroll
    for (int it = 0; it < 32; ++it) {
        int c  = threadIdx.x + it * 256;
        int kb = c >> 8;
        int rr = (c >> 2) & 63;
        int cc = c & 3;
        int grow = (rr >> 4) * H_ + n0 + (rr & 15);
        lds_cp16(WhT + (size_t)grow * H_ + kb * 32 + cc * 8,
                 smemB + kb * KB_ELEMS + rr * ROW_ELEMS + cc * 8);
    }
    lds_cp_wait();
    __syncthreads();

    const v8f vzero = {0.f, 0.f, 0.f, 0.f, 0.f, 0.f, 0.f, 0.f};
    v8f acc[4];
#pragma unroll
    for (int g = 0; g < 4; ++g) acc[g] = vzero;

    // A fragment base: row m0+l16; lanes 0-15 take K=[0..7,16..23], lanes 16-31 K=[8..15,24..31]
    const unsigned short* aBase = hprev + (size_t)(m0 + l16) * H_ + half * 8;
    // B per-lane base: row l16, K-half select; per iteration advance by one K-chunk
    const unsigned short* bPtr = smemB + l16 * ROW_ELEMS + half * 8;

    for (int kb = 0; kb < 32; ++kb) {
        V16U a;
        a.q[0] = *(const uint4*)(aBase + kb * 32);
        a.q[1] = *(const uint4*)(aBase + kb * 32 + 16);
        // load phase: all 4 B fragments into distinct registers (8 ds_loads in flight)
        V16U b[4];
#pragma unroll
        for (int g = 0; g < 4; ++g) {
            b[g].q[0] = *(const uint4*)(bPtr + g * GATE_ELEMS);
            b[g].q[1] = *(const uint4*)(bPtr + g * GATE_ELEMS + 16);
        }
        // compute phase
#pragma unroll
        for (int g = 0; g < 4; ++g) {
            acc[g] = __builtin_amdgcn_wmma_f32_16x16x32_bf16(
                false, a.v, false, b[g].v, (short)0, acc[g], false, false);
        }
        bPtr += KB_ELEMS;
    }

    V8U ai, af, ag, ao;
    ai.v = acc[0]; af.v = acc[1]; ag.v = acc[2]; ao.v = acc[3];

    const int nh = n0 + l16;  // h-column for this lane
#pragma unroll
    for (int r = 0; r < 8; ++r) {
        const int bb = m0 + half * 8 + r;                 // batch row
        const int tk = tok[bb * tokStride + t];
        const float* wrow = Wi + (size_t)tk * FOURH + nh; // one-hot gather
        float zi = ai.f[r] + wrow[0]          + bias[nh];
        float zf = af.f[r] + wrow[H_]         + bias[H_ + nh];
        float zg = ag.f[r] + wrow[2 * H_]     + bias[2 * H_ + nh];
        float zo = ao.f[r] + wrow[3 * H_]     + bias[3 * H_ + nh];
        float ig = sigmoidf_(zi);
        float fg = sigmoidf_(zf);
        float gg = tanhf(zg);
        float og = sigmoidf_(zo);
        float cp = cbuf[(size_t)bb * H_ + nh];
        float cn = fg * cp + ig * gg;
        float hn = og * tanhf(cn);
        cbuf[(size_t)bb * H_ + nh]  = cn;
        hnext[(size_t)bb * H_ + nh] = f2bf(hn);
        if (idx != nullptr && idx[bb] == t) {
            csel[(size_t)bb * H_ + nh] = cn;
            hsel[(size_t)bb * H_ + nh] = f2bf(hn);
        }
        if (hd != nullptr) {
            hd[((size_t)bb * Tdec + t) * H_ + nh] = f2bf(hn);
        }
    }
}

// ---------------------------------------------------------------------------
// Final dense: logits[M,V] = hd[M,H] @ dense_W[H,V] + b, all-WMMA.
// A tile (16 rows x K=1024, shared by all 8 waves) is staged in LDS;
// B = WdT bf16 [V,H] streamed from L2 (unique per wave).
// ---------------------------------------------------------------------------
__global__ void __launch_bounds__(256) dense_kernel(
    const unsigned short* __restrict__ A,    // [M, H] bf16
    const unsigned short* __restrict__ WdT,  // [V, H] bf16
    const float*          __restrict__ db,   // [V]
    float*                __restrict__ out,  // [M, V]
    int M)
{
    extern __shared__ unsigned short smemA[];     // [16][LDS_STRIDE]

    const int lane = threadIdx.x & 31;
    const int wave = threadIdx.x >> 5;
    const int half = lane >> 4;
    const int l16  = lane & 15;
    const int m0   = blockIdx.x * 16;
    const int nc0  = wave * 64;

    // ---- stage A tile: 16 rows x 128 b128-chunks = 2048 chunks, 8 per thread ----
#pragma unroll
    for (int it = 0; it < 8; ++it) {
        int c  = threadIdx.x + it * 256;
        int rr = c >> 7;
        int cc = (c & 127) << 3;
        lds_cp16(A + (size_t)(m0 + rr) * H_ + cc, smemA + rr * LDS_STRIDE + cc);
    }
    lds_cp_wait();
    __syncthreads();

    const v8f vzero = {0.f, 0.f, 0.f, 0.f, 0.f, 0.f, 0.f, 0.f};
    v8f acc[4];
#pragma unroll
    for (int j = 0; j < 4; ++j) acc[j] = vzero;

    const unsigned short* aRow = smemA + l16 * LDS_STRIDE + half * 8;

    for (int k0 = 0; k0 < H_; k0 += 32) {
        V16U a;
        a.q[0] = *(const uint4*)(aRow + k0);
        a.q[1] = *(const uint4*)(aRow + k0 + 16);
        // load phase: all 4 B fragments into distinct registers
        V16U b[4];
#pragma unroll
        for (int j = 0; j < 4; ++j) {
            const unsigned short* bp =
                WdT + (size_t)(nc0 + j * 16 + l16) * H_ + k0 + half * 8;
            b[j].q[0] = *(const uint4*)(bp);
            b[j].q[1] = *(const uint4*)(bp + 16);
        }
        // compute phase
#pragma unroll
        for (int j = 0; j < 4; ++j) {
            acc[j] = __builtin_amdgcn_wmma_f32_16x16x32_bf16(
                false, a.v, false, b[j].v, (short)0, acc[j], false, false);
        }
    }

#pragma unroll
    for (int j = 0; j < 4; ++j) {
        V8U d; d.v = acc[j];
        const int n = nc0 + j * 16 + l16;
        const float bn = db[n];
#pragma unroll
        for (int r = 0; r < 8; ++r) {
            const int m = m0 + half * 8 + r;
            out[(size_t)m * V_ + n] = d.f[r] + bn;
        }
    }
    (void)M;
}

// ---------------------------------------------------------------------------
// Host-side orchestration
// ---------------------------------------------------------------------------
extern "C" void kernel_launch(void* const* d_in, const int* in_sizes, int n_in,
                              void* d_out, int out_size, void* d_ws, size_t ws_size,
                              hipStream_t stream) {
    (void)in_sizes; (void)n_in; (void)out_size; (void)ws_size;

    const float* enc_in  = (const float*)d_in[0];
    const float* dec_in  = (const float*)d_in[1];
    const float* enc_Wi  = (const float*)d_in[2];
    const float* enc_Wh  = (const float*)d_in[3];
    const float* enc_b   = (const float*)d_in[4];
    const float* dec_Wi  = (const float*)d_in[5];
    const float* dec_Wh  = (const float*)d_in[6];
    const float* dec_b   = (const float*)d_in[7];
    const float* dense_W = (const float*)d_in[8];
    const float* dense_b = (const float*)d_in[9];
    float* out = (float*)d_out;

    // workspace carve-up (256B aligned slices)
    char* ws = (char*)d_ws;
    size_t off = 0;
    auto take = [&](size_t bytes) -> char* {
        char* p = ws + off;
        off = (off + bytes + 255) & ~(size_t)255;
        return p;
    };
    int* enc_tok = (int*)take((size_t)B_ * T_ * sizeof(int));
    int* dec_tok = (int*)take((size_t)B_ * T_ * sizeof(int));
    int* idxp    = (int*)take((size_t)B_ * sizeof(int));
    unsigned short* WhTe = (unsigned short*)take((size_t)FOURH * H_ * 2);
    unsigned short* WhTd = (unsigned short*)take((size_t)FOURH * H_ * 2);
    unsigned short* WdT  = (unsigned short*)take((size_t)V_ * H_ * 2);
    unsigned short* hA   = (unsigned short*)take((size_t)B_ * H_ * 2);
    unsigned short* hB   = (unsigned short*)take((size_t)B_ * H_ * 2);
    float*          cbuf = (float*)take((size_t)B_ * H_ * 4);
    float*          csel = (float*)take((size_t)B_ * H_ * 4);
    unsigned short* hsel = (unsigned short*)take((size_t)B_ * H_ * 2);
    unsigned short* hd   = (unsigned short*)take((size_t)B_ * (T_ - 1) * H_ * 2);

    const size_t lstmLds  = (size_t)32 * KB_ELEMS * sizeof(unsigned short);   // 163840 B
    const size_t denseLds = (size_t)16 * LDS_STRIDE * sizeof(unsigned short); //  33024 B

    // ---- prep ----
    tokenize_kernel<<<dim3((B_ * T_ + 255) / 256), dim3(256), 0, stream>>>(enc_in, enc_tok, B_ * T_);
    tokenize_kernel<<<dim3((B_ * T_ + 255) / 256), dim3(256), 0, stream>>>(dec_in, dec_tok, B_ * T_);
    seqidx_kernel<<<dim3(1), dim3(128), 0, stream>>>(enc_tok, idxp);
    transpose_f32_to_bf16<<<dim3(FOURH / 32, H_ / 32), dim3(32, 8), 0, stream>>>(enc_Wh, WhTe, H_, FOURH);
    transpose_f32_to_bf16<<<dim3(FOURH / 32, H_ / 32), dim3(32, 8), 0, stream>>>(dec_Wh, WhTd, H_, FOURH);
    transpose_f32_to_bf16<<<dim3(V_ / 32, H_ / 32), dim3(32, 8), 0, stream>>>(dense_W, WdT, H_, V_);
    init_state_kernel<<<dim3((B_ * H_ + 255) / 256), dim3(256), 0, stream>>>(cbuf, hA, B_ * H_);

    // ---- encoder scan ----
    unsigned short* cur = hA;
    unsigned short* nxt = hB;
    for (int t = 0; t < T_; ++t) {
        lstm_step_kernel<<<dim3(H_ / 16), dim3(256), lstmLds, stream>>>(
            WhTe, enc_Wi, enc_b, enc_tok, T_, t, cur, cbuf, nxt,
            idxp, csel, hsel, (unsigned short*)nullptr, 0);
        unsigned short* tmp = cur; cur = nxt; nxt = tmp;
    }

    // ---- carry selected encoder state into decoder init ----
    copy_state_kernel<<<dim3((B_ * H_ + 255) / 256), dim3(256), 0, stream>>>(csel, hsel, cbuf, hA, B_ * H_);

    // ---- decoder scan (T-1 steps over decoder_inputs[:, :-1]) ----
    cur = hA; nxt = hB;
    for (int t = 0; t < T_ - 1; ++t) {
        lstm_step_kernel<<<dim3(H_ / 16), dim3(256), lstmLds, stream>>>(
            WhTd, dec_Wi, dec_b, dec_tok, T_, t, cur, cbuf, nxt,
            (const int*)nullptr, (float*)nullptr, (unsigned short*)nullptr, hd, T_ - 1);
        unsigned short* tmp = cur; cur = nxt; nxt = tmp;
    }

    // ---- final dense projection ----
    dense_kernel<<<dim3((B_ * (T_ - 1)) / 16), dim3(256), denseLds, stream>>>(hd, WdT, dense_b, out, B_ * (T_ - 1));
}
